// LSTM_47356309405872
// MI455X (gfx1250) — matrix operand order repaired and checked by reading, not compiled, not run on hardware
//
#include <hip/hip_runtime.h>
#include <cstdint>
#include <cstddef>

// ---------------- problem constants ----------------
#define S_LEN 512
#define BATCH 64
#define HID   1024
#define GATES 4096      // 4*HID
#define NWG   64        // persistent recurrent workgroups (one WGP each)
#define WLD   1032      // LDS row stride for weight slice: 2064B = 16*129 (aligned, 4-bank skew)

typedef __bf16        v16bf __attribute__((ext_vector_type(16)));
typedef float         v8f   __attribute__((ext_vector_type(8)));
typedef unsigned int  u32x4 __attribute__((ext_vector_type(4)));

union FragU { u32x4 u[2]; v16bf v; };

__device__ __forceinline__ v8f wmma_bf16(v16bf a, v16bf b, v8f c) {
  // (neg_a, A, neg_b, B, c_mod, C, reuse_a, reuse_b) -> v_wmma_f32_16x16x32_bf16
  return __builtin_amdgcn_wmma_f32_16x16x32_bf16(false, a, false, b, (short)0, c, false, false);
}

// -------- async global->LDS copy (CDNA5 GLOBAL_LOAD_ASYNC_TO_LDS_B128, ASYNCcnt) --------
#if defined(__has_builtin)
#if __has_builtin(__builtin_amdgcn_global_load_async_to_lds_b128)
#define ASYNC_CP 1
#endif
#endif

#if ASYNC_CP
typedef int v4i __attribute__((ext_vector_type(4)));
// NOTE: attribute in typedef binds to the pointee type (cast-position attributes get dropped)
typedef __attribute__((address_space(1))) v4i* gv4i_ptr;
typedef __attribute__((address_space(3))) v4i* lv4i_ptr;
__device__ __forceinline__ void cp_async_b128(void* lds, const void* g) {
  __builtin_amdgcn_global_load_async_to_lds_b128((gv4i_ptr)g, (lv4i_ptr)lds, 0, 0);
}
__device__ __forceinline__ void cp_async_wait() {
  asm volatile("s_wait_asynccnt 0x0" ::: "memory");
}
#else
__device__ __forceinline__ void cp_async_b128(void* lds, const void* g) {
  *(u32x4*)lds = *(const u32x4*)g;           // fallback: VGPR round-trip
}
__device__ __forceinline__ void cp_async_wait() {
  asm volatile("" ::: "memory");
}
#endif

// A fragment 16x32 (MxK), row-major source (K contiguous).
// ISA layout: lane<16: M=lane, K={0..7, 16..23}; lane>=16: M=lane-16, K={8..15, 24..31}
__device__ __forceinline__ v16bf load_frag_a(const __bf16* p, int row0, int k0, int ld, int lane) {
  int m  = lane & 15;
  int kb = (lane >> 4) << 3;                 // 0 or 8
  const __bf16* r = p + (size_t)(row0 + m) * ld + (k0 + kb);
  FragU f;
  f.u[0] = *(const u32x4*)(r);               // K = kb .. kb+7
  f.u[1] = *(const u32x4*)(r + 16);          // K = kb+16 .. kb+23
  return f.v;
}

// B fragment 32x16 (KxN). B column n == row g of W (W stored [g][k], K contiguous).
// ISA layout: lane<16: N=lane, K={0..7, 16..23}; lane>=16: N=lane-16, K={8..15, 24..31}
__device__ __forceinline__ v16bf load_frag_b(const __bf16* p, int col0, int k0, int ld, int lane) {
  int n  = lane & 15;
  int kb = (lane >> 4) << 3;                 // 0 or 8
  const __bf16* r = p + (size_t)(col0 + n) * ld + (k0 + kb);
  FragU f;
  f.u[0] = *(const u32x4*)(r);
  f.u[1] = *(const u32x4*)(r + 16);
  return f.v;
}

__device__ __forceinline__ float sigm(float x) { return 1.0f / (1.0f + __expf(-x)); }

// ---------------- fp32 -> bf16 convert ----------------
__global__ void k_cvt_bf16(const float* __restrict__ s, __bf16* __restrict__ d, int n) {
  int i = blockIdx.x * blockDim.x + threadIdx.x;
  int stride = gridDim.x * blockDim.x;
  for (; i < n; i += stride) d[i] = (__bf16)s[i];
}

// ---------------- input projection GEMM ----------------
// C[M][4096] = A[M][1024](bf16) * W[4096][1024]^T(bf16) + (b_ih + b_hh), fp32 out.
// 256 thr (8 waves, 4x2), BM=BN=128 BK=32; wave tile 32x64 (2x4 WMMA, 6 frag loads / 8 wmma).
// Double-buffered LDS filled by async global->LDS copies: one barrier per k-step.
#define GLD 40          // 80B LDS row stride: 16B aligned, conflict-free fragment reads
__global__ __launch_bounds__(256) void k_gemm_gi(
    const __bf16* __restrict__ A, const __bf16* __restrict__ W,
    const float* __restrict__ bih, const float* __restrict__ bhh,
    float* __restrict__ C, int M) {
  __shared__ __bf16 As[2][128][GLD];
  __shared__ __bf16 Bs[2][128][GLD];
  const int tid = threadIdx.x, lane = tid & 31, wave = tid >> 5;
  const int wm = wave >> 1, wn = wave & 1;           // 4x2 wave grid
  const int m0 = blockIdx.y * 128, g0 = blockIdx.x * 128;
  const int r = tid >> 1, kk = (tid & 1) * 16;       // this thread's slice of the 128x32 tiles

  auto issue = [&](int kt, int buf) {
    const __bf16* asrc = A + (size_t)(m0 + r) * HID + kt * 32 + kk;
    cp_async_b128(&As[buf][r][kk],     asrc);
    cp_async_b128(&As[buf][r][kk + 8], asrc + 8);
    const __bf16* bsrc = W + (size_t)(g0 + r) * HID + kt * 32 + kk;
    cp_async_b128(&Bs[buf][r][kk],     bsrc);
    cp_async_b128(&Bs[buf][r][kk + 8], bsrc + 8);
  };

  const v8f z = {0, 0, 0, 0, 0, 0, 0, 0};
  v8f acc[2][4] = {{z, z, z, z}, {z, z, z, z}};
  issue(0, 0);
  for (int kt = 0; kt < 32; ++kt) {
    cp_async_wait();                // own async copies landed in LDS
    __syncthreads();                // everyone's landed; prev buffer reads all done
    if (kt < 31) issue(kt + 1, (kt + 1) & 1);
    const __bf16* a = &As[kt & 1][0][0];
    const __bf16* b = &Bs[kt & 1][0][0];
    v16bf af[2], bf[4];
#pragma unroll
    for (int tm = 0; tm < 2; ++tm) af[tm] = load_frag_a(a, wm * 32 + tm * 16, 0, GLD, lane);
#pragma unroll
    for (int tn = 0; tn < 4; ++tn) bf[tn] = load_frag_b(b, wn * 64 + tn * 16, 0, GLD, lane);
#pragma unroll
    for (int tm = 0; tm < 2; ++tm)
#pragma unroll
      for (int tn = 0; tn < 4; ++tn)
        acc[tm][tn] = wmma_bf16(af[tm], bf[tn], acc[tm][tn]);
  }
  const int n = lane & 15, mh = (lane >> 4) * 8;
#pragma unroll
  for (int tm = 0; tm < 2; ++tm)
#pragma unroll
    for (int tn = 0; tn < 4; ++tn) {
      int gc = g0 + wn * 64 + tn * 16 + n;
      float bias = bih[gc] + bhh[gc];
#pragma unroll
      for (int j = 0; j < 8; ++j) {
        size_t row = (size_t)(m0 + wm * 32 + tm * 16 + mh + j);
        __builtin_nontemporal_store(acc[tm][tn][j] + bias, &C[row * GATES + gc]);
      }
    }
}

// ---------------- persistent recurrent layer ----------------
// 64 WGs; WG owns 16 columns of H (x4 gates = 64 gate cols). W_hh slice (64x1024
// bf16 = 128KB+pad) is LDS-resident for all 512 steps. c stays in registers.
// Per-step grid barrier: release atomicAdd + per-wave acquire spin.
__global__ __launch_bounds__(256) void k_lstm_rec(
    const float* __restrict__ gi,      // [S][B][4H] fp32
    const __bf16* __restrict__ whh,    // [4H][H] bf16
    const __bf16* __restrict__ h0z,    // zeros [B][H]
    __bf16* __restrict__ ybf,          // [S][B][H] bf16 out (also h chain)
    float* __restrict__ yf32,          // [S][B][H] fp32 out or nullptr
    float* __restrict__ hN, float* __restrict__ cN,   // [B][H]
    int* __restrict__ bar) {           // [S] step barrier counters (pre-zeroed)
  extern __shared__ __bf16 wlds[];     // [64][WLD]
  __shared__ float gl[BATCH][68];      // gate exchange, 4-bank row skew
  const int tid = threadIdx.x, lane = tid & 31, wave = tid >> 5;
  const int wm = wave >> 2, q = wave & 3;        // wave = (m-half, gate)
  const int n0 = blockIdx.x * 16;
  // Stage weight slice via async copies: LDS row r = q*16+j <-> global g = q*H + n0 + j
  for (int e = tid * 8; e < 64 * HID; e += 256 * 8) {
    int rr = e >> 10, k = e & 1023;
    int g = (rr >> 4) * HID + n0 + (rr & 15);
    cp_async_b128(&wlds[rr * WLD + k], whh + (size_t)g * HID + k);
  }
  cp_async_wait();
  __syncthreads();
  float creg[4] = {0.f, 0.f, 0.f, 0.f};
  const v8f z = {0, 0, 0, 0, 0, 0, 0, 0};
  for (int t = 0; t < S_LEN; ++t) {
    const __bf16* hp = (t == 0) ? h0z : (ybf + (size_t)(t - 1) * BATCH * HID);
    v8f acc0 = z, acc1 = z;
#pragma unroll 4
    for (int k0 = 0; k0 < HID; k0 += 32) {
      v16bf a0 = load_frag_a(hp, wm * 32,      k0, HID, lane);
      v16bf a1 = load_frag_a(hp, wm * 32 + 16, k0, HID, lane);
      v16bf b  = load_frag_b(wlds, q * 16, k0, WLD, lane);
      acc0 = wmma_bf16(a0, b, acc0);
      acc1 = wmma_bf16(a1, b, acc1);
    }
    const float* git = gi + (size_t)t * BATCH * GATES;
    const int n = lane & 15, mh = (lane >> 4) * 8;
    const int gc = q * HID + n0 + n;
#pragma unroll
    for (int j = 0; j < 8; ++j) {
      int b0 = wm * 32 + mh + j;
      gl[b0][q * 16 + n] = acc0[j] + __builtin_nontemporal_load(&git[(size_t)b0 * GATES + gc]);
      int b1 = b0 + 16;
      gl[b1][q * 16 + n] = acc1[j] + __builtin_nontemporal_load(&git[(size_t)b1 * GATES + gc]);
    }
    __syncthreads();
#pragma unroll
    for (int rr = 0; rr < 4; ++rr) {
      int e = tid + rr * 256;
      int b = e >> 4, j = e & 15;
      float ig = sigm(gl[b][j]);
      float fg = sigm(gl[b][16 + j]);
      float gg = tanhf(gl[b][32 + j]);
      float og = sigm(gl[b][48 + j]);
      float cnew = fg * creg[rr] + ig * gg;
      float hnew = og * tanhf(cnew);
      creg[rr] = cnew;
      size_t oi = (size_t)t * BATCH * HID + (size_t)b * HID + n0 + j;
      ybf[oi] = (__bf16)hnew;
      if (yf32) __builtin_nontemporal_store(hnew, &yf32[oi]);
      if (t == S_LEN - 1) {
        hN[b * HID + n0 + j] = hnew;
        cN[b * HID + n0 + j] = cnew;
      }
    }
    __syncthreads();
    if (tid == 0) {
      __threadfence();
      __hip_atomic_fetch_add(&bar[t], 1, __ATOMIC_RELEASE, __HIP_MEMORY_SCOPE_AGENT);
    }
    if ((tid & 31) == 0) {     // one acquire spinner per wave -> L0 invalidated on every CU
      while (__hip_atomic_load(&bar[t], __ATOMIC_ACQUIRE, __HIP_MEMORY_SCOPE_AGENT) < NWG)
        __builtin_amdgcn_s_sleep(2);
    }
    __syncthreads();
  }
}

// ---------------- host side ----------------
extern "C" void kernel_launch(void* const* d_in, const int* in_sizes, int n_in,
                              void* d_out, int out_size, void* d_ws, size_t ws_size,
                              hipStream_t stream) {
  (void)in_sizes; (void)n_in; (void)out_size; (void)ws_size;
  const float* x    = (const float*)d_in[0];
  const float* wih0 = (const float*)d_in[1];
  const float* whh0 = (const float*)d_in[2];
  const float* bih0 = (const float*)d_in[3];
  const float* bhh0 = (const float*)d_in[4];
  const float* wih1 = (const float*)d_in[5];
  const float* whh1 = (const float*)d_in[6];
  const float* bih1 = (const float*)d_in[7];
  const float* bhh1 = (const float*)d_in[8];
  float* out = (float*)d_out;

  const size_t nX = (size_t)S_LEN * BATCH * HID;   // 33.5M
  const size_t nW = (size_t)GATES * HID;           // 4.2M
  char* ws = (char*)d_ws;
  size_t o = 0;
  auto take = [&](size_t bytes) { char* p = ws + o; o += (bytes + 255) & ~(size_t)255; return p; };
  __bf16* xb    = (__bf16*)take(nX * 2);
  __bf16* wih0b = (__bf16*)take(nW * 2);
  __bf16* whh0b = (__bf16*)take(nW * 2);
  __bf16* wih1b = (__bf16*)take(nW * 2);
  __bf16* whh1b = (__bf16*)take(nW * 2);
  __bf16* y0b   = (__bf16*)take(nX * 2);
  __bf16* y1b   = (__bf16*)take(nX * 2);
  __bf16* zb    = (__bf16*)take((size_t)BATCH * HID * 2);
  float*  gi    = (float*)take((size_t)S_LEN * BATCH * GATES * 4);
  int*    bar0  = (int*)take(S_LEN * sizeof(int));
  int*    bar1  = (int*)take(S_LEN * sizeof(int));

  // fp32 -> bf16 operand conversion (redone every call: deterministic)
  k_cvt_bf16<<<2048, 256, 0, stream>>>(x,    xb,    (int)nX);
  k_cvt_bf16<<<1024, 256, 0, stream>>>(wih0, wih0b, (int)nW);
  k_cvt_bf16<<<1024, 256, 0, stream>>>(whh0, whh0b, (int)nW);
  k_cvt_bf16<<<1024, 256, 0, stream>>>(wih1, wih1b, (int)nW);
  k_cvt_bf16<<<1024, 256, 0, stream>>>(whh1, whh1b, (int)nW);
  (void)hipMemsetAsync(zb, 0, (size_t)BATCH * HID * 2, stream);
  (void)hipMemsetAsync(bar0, 0, S_LEN * sizeof(int), stream);
  (void)hipMemsetAsync(bar1, 0, S_LEN * sizeof(int), stream);

  float* hn = out + nX;                    // h_n [2][B][H]
  float* cn = hn + 2 * BATCH * HID;        // c_n [2][B][H]

  const int REC_SMEM = 64 * WLD * 2;       // 132,096 B dynamic LDS (+17KB static, <320KB/WGP)
  (void)hipFuncSetAttribute((const void*)k_lstm_rec,
                            hipFuncAttributeMaxDynamicSharedMemorySize, REC_SMEM);

  dim3 gg(GATES / 128, (S_LEN * BATCH) / 128);
  // Layer 0
  k_gemm_gi<<<gg, 256, 0, stream>>>(xb, wih0b, bih0, bhh0, gi, S_LEN * BATCH);
  k_lstm_rec<<<NWG, 256, REC_SMEM, stream>>>(gi, whh0b, zb, y0b, nullptr, hn, cn, bar0);
  // Layer 1 (gi buffer reused; y1 written fp32 straight into d_out)
  k_gemm_gi<<<gg, 256, 0, stream>>>(y0b, wih1b, bih1, bhh1, gi, S_LEN * BATCH);
  k_lstm_rec<<<NWG, 256, REC_SMEM, stream>>>(gi, whh1b, zb, y1b, out,
                                             hn + BATCH * HID, cn + BATCH * HID, bar1);
}
